// IALSHAttention16_46299747451199
// MI455X (gfx1250) — matrix operand order repaired and checked by reading, not compiled, not run on hardware
//
#include <hip/hip_runtime.h>
#include <hip/hip_bf16.h>
#include <math.h>

// Problem constants from the reference
#define BB   4
#define HH   8
#define SS   2048
#define DD   64
#define DP   66     // D + M_AUG
#define KP   96     // K padded to 3 x 32 for wmma_f32_16x16x32_f16
#define UCONST 0.75f
#define TOPK 32
#define NR   (BB*HH*SS)   // 65536 rows

typedef __attribute__((ext_vector_type(16))) _Float16 v16h;
typedef __attribute__((ext_vector_type(8)))  float    v8f;

// ---------------- workspace layout (floats) ----------------
// [0] = Mbits, [1] = M2bits, pad to 16 floats
// narr  : NR floats
// qdot  : NR floats
// Pnorm : NR floats
// Qs    : NR floats
// Pfin  : NR * KP halves  (fp16, rows padded to 96)
#define WS_SCAL 16
#define WS_N    (WS_SCAL)
#define WS_QD   (WS_N + NR)
#define WS_PN   (WS_QD + NR)
#define WS_QS   (WS_PN + NR)
#define WS_PF   (WS_QS + NR)   // float offset; cast to _Float16* from here

__global__ void k_zero_scalars(unsigned* s) {
    if (threadIdx.x < 2) s[threadIdx.x] = 0u;
}

// Per row: ||qk||, dot(qk, a[:64]); global max ||qk|| via uint atomicMax
__global__ __launch_bounds__(256) void k_norm_dot(
    const _Float16* __restrict__ qk, const _Float16* __restrict__ a,
    float* __restrict__ narr, float* __restrict__ qdot, unsigned* __restrict__ Mbits)
{
    int r = blockIdx.x * blockDim.x + threadIdx.x;
    if (r >= NR) return;
    const _Float16* q  = qk + (size_t)r * DD;
    const _Float16* av = a  + (size_t)r * DP;
    float ss = 0.f, dt = 0.f;
    #pragma unroll
    for (int d = 0; d < DD; ++d) {
        float x = (float)q[d];
        ss += x * x;
        dt += x * (float)av[d];
    }
    float n = sqrtf(ss);
    narr[r] = n;
    qdot[r] = dt;
    atomicMax(Mbits, __float_as_uint(n));   // positive floats order as uints
}

// Per row: P_norm (uses pre-scale norm for augmented terms, per reference), max P_norm
__global__ __launch_bounds__(256) void k_pnorm(
    const float* __restrict__ narr, float* __restrict__ Pnorm,
    const unsigned* __restrict__ Mbits, unsigned* __restrict__ M2bits)
{
    int r = blockIdx.x * blockDim.x + threadIdx.x;
    if (r >= NR) return;
    float M  = __uint_as_float(*Mbits);
    float s  = UCONST / M;
    float n  = narr[r];
    float ns = n * s;
    float n2 = n * n;
    float e1 = 0.5f - n2;
    float e2 = 0.5f - n2 * n2;
    float pn = sqrtf(ns * ns + e1 * e1 + e2 * e2);
    Pnorm[r] = pn;
    atomicMax(M2bits, __float_as_uint(pn));
}

// Per row: materialize P_final (fp16, padded to KP) and Qs
__global__ __launch_bounds__(256) void k_pfin(
    const _Float16* __restrict__ qk,
    const float* __restrict__ narr, const float* __restrict__ qdot,
    const float* __restrict__ Pnorm,
    const unsigned* __restrict__ Mbits, const unsigned* __restrict__ M2bits,
    _Float16* __restrict__ Pfin, float* __restrict__ Qs)
{
    int r = blockIdx.x * blockDim.x + threadIdx.x;
    if (r >= NR) return;
    float M  = __uint_as_float(*Mbits);
    float M2 = __uint_as_float(*M2bits);
    float s  = UCONST / M;
    float n  = narr[r];
    float pn = Pnorm[r];
    float f  = M2 / pn;
    const _Float16* q = qk + (size_t)r * DD;
    _Float16* p = Pfin + (size_t)r * KP;
    #pragma unroll
    for (int d = 0; d < DD; ++d) p[d] = (_Float16)((float)q[d] * s * f);
    float n2 = n * n;
    p[DD + 0] = (_Float16)((0.5f - n2) * f);
    p[DD + 1] = (_Float16)((0.5f - n2 * n2) * f);
    #pragma unroll
    for (int d = DD + 2; d < KP; ++d) p[d] = (_Float16)0.0f;
    Qs[r] = (M2 / n) * qdot[r];   // Q_final = qk/n * M2
}

// Main fused kernel: per (bh, 16-row j-tile)
//   scores[j,i] = (a[j,:] . Pfin[i,:]) * Qs[i]   via v_wmma_f32_16x16x32_f16
//   top-32 per j row over 2048 i's; write fp32 mask (0 at top-k, -10000 else)
__global__ __launch_bounds__(256) void k_scores_topk(
    const _Float16* __restrict__ a, const _Float16* __restrict__ Pfin,
    const float* __restrict__ Qs, float* __restrict__ out)
{
    __shared__ _Float16 sc[16 * SS];   // 64 KB score tile (fp16)

    const int bh  = blockIdx.y;
    const int j0  = blockIdx.x * 16;
    const int tid  = threadIdx.x;
    const int wv   = tid >> 5;         // 8 waves
    const int lane = tid & 31;
    const int g  = lane >> 4;          // k-group for wave32 fragment layout
    const int mr = lane & 15;          // A: row-in-tile; B: col-in-tile

    // ---- A fragments: a[j0+mr, :] padded 66 -> 96, wave32 16x32 f16 layout ----
    const _Float16* arow = a + ((size_t)(bh * SS + j0 + mr)) * DP;
    v16h afrag[3];
    #pragma unroll
    for (int kb = 0; kb < 3; ++kb) {
        #pragma unroll
        for (int h = 0; h < 8; ++h) {
            int k1 = kb * 32 + g * 16 + h;
            int k2 = kb * 32 + g * 16 + 8 + h;
            afrag[kb][h]     = (k1 < DP) ? arow[k1] : (_Float16)0.0f;
            afrag[kb][h + 8] = (k2 < DP) ? arow[k2] : (_Float16)0.0f;
        }
    }

    // ---- score tile: each wave owns i-tiles t = wv, wv+8, ... (columns disjoint) ----
    const _Float16* Pbh = Pfin + (size_t)bh * SS * KP;
    for (int t = wv; t < SS / 16; t += 8) {
        int i0 = t * 16;
        const _Float16* prow = Pbh + (size_t)(i0 + mr) * KP + g * 16;  // 32B aligned
        __builtin_prefetch(prow + 8 * 16 * KP, 0, 1);                  // next tile stream
        v8f c = {};
        #pragma unroll
        for (int kb = 0; kb < 3; ++kb) {
            v16h bfrag = *(const v16h*)(prow + kb * 32);
            c = __builtin_amdgcn_wmma_f32_16x16x32_f16(
                    false, afrag[kb], false, bfrag, (short)0, c, false, false);
        }
        float qsv = Qs[bh * SS + i0 + mr];   // column scale Qs[i]
        #pragma unroll
        for (int r8 = 0; r8 < 8; ++r8) {
            float v = c[r8] * qsv;
            v = (v == v) ? v : 0.0f;         // NaN -> 0 per reference
            sc[(r8 + 8 * g) * SS + i0 + mr] = (_Float16)v;
        }
    }
    __syncthreads();

    // ---- top-32 per row: iterative argmax extraction, winner marks -inf in LDS ----
    const float NEG_INF = -__builtin_huge_valf();
    for (int rr = 0; rr < 2; ++rr) {
        int m = wv * 2 + rr;                 // row of the tile owned by this wave
        _Float16* row = sc + m * SS;
        for (int it = 0; it < TOPK; ++it) {
            float bm = -3.0e38f;
            int   be = 0;
            for (int e = 0; e < 64; ++e) {   // lane owns i = e*32 + lane (bank-free)
                float v = (float)row[e * 32 + lane];
                if (v > bm) { bm = v; be = e; }
            }
            unsigned key = ((unsigned)lane << 8) | (unsigned)be;
            #pragma unroll
            for (int off = 16; off; off >>= 1) {
                float    ov = __shfl_xor(bm, off, 32);
                unsigned ok = (unsigned)__shfl_xor((int)key, off, 32);
                if (ov > bm || (ov == bm && ok < key)) { bm = ov; key = ok; }
            }
            if ((key >> 8) == (unsigned)lane) {
                row[(key & 255u) * 32 + lane] = (_Float16)NEG_INF;  // extract
            }
            __syncthreads();                 // uniform across all waves (2*32 iters)
        }
        // ---- write mask row: 0 where extracted (-inf), -10000 elsewhere ----
        float* orow = out + ((size_t)bh * SS + (size_t)(j0 + m)) * SS;
        for (int e = 0; e < 64; ++e) {
            int i = e * 32 + lane;
            float v = (float)row[i];
            orow[i] = (v == NEG_INF) ? 0.0f : -10000.0f;
        }
    }
}

extern "C" void kernel_launch(void* const* d_in, const int* in_sizes, int n_in,
                              void* d_out, int out_size, void* d_ws, size_t ws_size,
                              hipStream_t stream) {
    const _Float16* qk = (const _Float16*)d_in[0];
    const _Float16* a  = (const _Float16*)d_in[1];
    float* out = (float*)d_out;

    float*     wsf   = (float*)d_ws;
    unsigned*  scal  = (unsigned*)d_ws;
    float*     narr  = wsf + WS_N;
    float*     qdot  = wsf + WS_QD;
    float*     Pnorm = wsf + WS_PN;
    float*     Qs    = wsf + WS_QS;
    _Float16*  Pfin  = (_Float16*)(wsf + WS_PF);

    k_zero_scalars<<<1, 32, 0, stream>>>(scal);
    k_norm_dot<<<NR / 256, 256, 0, stream>>>(qk, a, narr, qdot, scal);
    k_pnorm   <<<NR / 256, 256, 0, stream>>>(narr, Pnorm, scal, scal + 1);
    k_pfin    <<<NR / 256, 256, 0, stream>>>(qk, narr, qdot, Pnorm, scal, scal + 1, Pfin, Qs);

    dim3 grid(SS / 16, BB * HH);
    k_scores_topk<<<grid, 256, 0, stream>>>(a, Pfin, Qs, out);
}
